// HGPSLPool_21079699489372
// MI455X (gfx1250) — matrix operand order repaired and checked by reading, not compiled, not run on hardware
//
#include <hip/hip_runtime.h>
#include <hip/hip_bf16.h>

// Problem constants (from reference: B=128, NPG=512, C=128, EPG=8192)
#define NGRAPH   128
#define NPG      512
#define CCH      128
#define NNODES   (NGRAPH * NPG)          // 65536
#define EEDGES   (NGRAPH * 8192)         // 1048576
#define KPOOL    410                     // ceil(0.8*512)
#define NK       (NGRAPH * KPOOL)        // 52480
#define SLOPE    0.2f
#define LAMB     1.0f
#define NITER    13                      // ceil(410/32)

typedef __attribute__((ext_vector_type(16))) _Float16 v16h;
typedef __attribute__((ext_vector_type(8)))  float    v8f;

// ---------------- generic fills ----------------
__global__ void hgpsl_fill_f32(float* p, float v, int n) {
    for (int i = blockIdx.x * blockDim.x + threadIdx.x; i < n;
         i += gridDim.x * blockDim.x) p[i] = v;
}
__global__ void hgpsl_fill_i32(int* p, int v, int n) {
    for (int i = blockIdx.x * blockDim.x + threadIdx.x; i < n;
         i += gridDim.x * blockDim.x) p[i] = v;
}

// ---------------- 1. degree scatter (by source row) ----------------
__global__ void hgpsl_deg(const int* __restrict__ row, const float* __restrict__ ea,
                          float* __restrict__ deg) {
    int e = blockIdx.x * blockDim.x + threadIdx.x;
    if (e < EEDGES) atomicAdd(&deg[row[e]], ea[e]);
}

// deg -> D^{-1/2} in place
__global__ void hgpsl_dis(float* deg) {
    int i = blockIdx.x * blockDim.x + threadIdx.x;
    if (i >= NNODES) return;
    float d = deg[i];
    deg[i] = (d > 0.f) ? rsqrtf(fmaxf(d, 1e-12f)) : 0.f;
}

// ---------------- 2. agg[col] += coef * x[row] (4 channels / thread) ----------------
__global__ void hgpsl_agg(const int* __restrict__ row, const int* __restrict__ col,
                          const float* __restrict__ ea, const float* __restrict__ dis,
                          const float* __restrict__ x, float* __restrict__ agg) {
    unsigned t = blockIdx.x * blockDim.x + threadIdx.x;   // EEDGES*32 threads
    unsigned e = t >> 5;
    if (e >= EEDGES) return;
    int cg = (int)(t & 31u) << 2;
    int r = row[e], c = col[e];
    float coef = dis[r] * ea[e] * dis[c];
    float4 xv = *reinterpret_cast<const float4*>(x + (size_t)r * CCH + cg);
    float* dst = agg + (size_t)c * CCH + cg;
    atomicAdd(dst + 0, coef * xv.x);
    atomicAdd(dst + 1, coef * xv.y);
    atomicAdd(dst + 2, coef * xv.z);
    atomicAdd(dst + 3, coef * xv.w);
}

// ---------------- 3. score = sum_c |x - agg| ----------------
__global__ void hgpsl_score(const float* __restrict__ x, const float* __restrict__ agg,
                            float* __restrict__ score) {
    int i = blockIdx.x * blockDim.x + threadIdx.x;
    if (i >= NNODES) return;
    const float4* xv = reinterpret_cast<const float4*>(x + (size_t)i * CCH);
    const float4* av = reinterpret_cast<const float4*>(agg + (size_t)i * CCH);
    float s = 0.f;
#pragma unroll 8
    for (int j = 0; j < CCH / 4; ++j) {
        float4 a = xv[j], b = av[j];
        s += fabsf(a.x - b.x) + fabsf(a.y - b.y) + fabsf(a.z - b.z) + fabsf(a.w - b.w);
    }
    score[i] = s;
}

// ---------------- 4. per-graph bitonic top-k (512 elements, 512 threads) ----------------
__global__ __launch_bounds__(NPG) void hgpsl_topk(const float* __restrict__ score,
                                                  int* __restrict__ perm,
                                                  int* __restrict__ mapping,
                                                  float* __restrict__ out_batch) {
    __shared__ float ss[NPG];
    __shared__ int   si[NPG];
    int g = blockIdx.x, t = threadIdx.x;
    ss[t] = score[g * NPG + t];
    si[t] = t;
    for (int size = 2; size <= NPG; size <<= 1) {
        for (int stride = size >> 1; stride > 0; stride >>= 1) {
            __syncthreads();
            int j = t ^ stride;
            if (j > t) {
                float a = ss[t], b = ss[j];
                int ia = si[t], ib = si[j];
                bool aFirst  = (a > b) || (a == b && ia < ib);  // descending, stable-ish
                bool dirDesc = ((t & size) == 0);
                if (aFirst != dirDesc) { ss[t] = b; ss[j] = a; si[t] = ib; si[j] = ia; }
            }
        }
    }
    __syncthreads();
    if (t < KPOOL) {
        int node = g * NPG + si[t];
        int pi   = g * KPOOL + t;
        perm[pi] = node;
        mapping[node] = pi;
        out_batch[pi] = (float)g;   // batch_pool (value-cast int->float)
    }
}

// ---------------- 5. gather x_pool ----------------
__global__ void hgpsl_gather(const float* __restrict__ x, const int* __restrict__ perm,
                             float* __restrict__ outx) {
    int i = blockIdx.x * blockDim.x + threadIdx.x;     // NK * 32
    if (i >= NK * (CCH / 4)) return;
    int p = i / (CCH / 4);
    int q = i % (CCH / 4);
    float4 v = reinterpret_cast<const float4*>(x + (size_t)perm[p] * CCH)[q];
    reinterpret_cast<float4*>(outx + (size_t)p * CCH)[q] = v;
}

// ---------------- 6. induced adjacency scatter ----------------
__global__ void hgpsl_induced(const int* __restrict__ row, const int* __restrict__ col,
                              const float* __restrict__ ea, const int* __restrict__ mapping,
                              float* __restrict__ induced) {
    int e = blockIdx.x * blockDim.x + threadIdx.x;
    if (e >= EEDGES) return;
    int mr = mapping[row[e]], mc = mapping[col[e]];
    if (mr >= 0 && mc >= 0)
        atomicAdd(&induced[(size_t)mr * KPOOL + (mc % KPOOL)], ea[e]);
}

// ---------------- 7. e1/e2 = x_pool . att via WMMA (one wave / 16 rows) ----------------
// A: 16x32 f16 tile of x_pool; B: 32x16 with col0 = att[0:C], col1 = att[C:2C], rest 0.
// Branchless B build: pointer-select att half per lane + 0/1 zero-mask, 4x float4 loads.
// D(f32 16x16): col0 -> e1, col1 -> e2. NK % 16 == 0, grid exact -> EXEC all ones.
__global__ __launch_bounds__(256) void hgpsl_ewmma(const float* __restrict__ xp,
                                                   const float* __restrict__ att,
                                                   float* __restrict__ e1,
                                                   float* __restrict__ e2) {
    int wave = (blockIdx.x * blockDim.x + threadIdx.x) >> 5;  // tile id
    int lane = threadIdx.x & 31;
    int m    = lane & 15;
    int half = lane >> 4;
    int row  = wave * 16 + m;

    const float* battp = att + ((m == 1) ? CCH : 0);   // lanes m>=2 read att[0:..], zeroed
    float zmask = (m < 2) ? 1.f : 0.f;

    v8f acc = {0.f, 0.f, 0.f, 0.f, 0.f, 0.f, 0.f, 0.f};
#pragma unroll
    for (int ks = 0; ks < CCH; ks += 32) {
        v16h a, b;
        // A fragment: lane(m,half); VGPR j<4 -> K=2j+8*half ; j>=4 -> K=16+2(j-4)+8*half
#pragma unroll
        for (int j = 0; j < 8; ++j) {
            int kk = (j < 4) ? (2 * j + 8 * half) : (16 + 2 * (j - 4) + 8 * half);
            float2 xv = *reinterpret_cast<const float2*>(xp + (size_t)row * CCH + ks + kk);
            a[2 * j]     = (_Float16)xv.x;
            a[2 * j + 1] = (_Float16)xv.y;
        }
        // B fragment: N = m, K = ks + 8*half + t (t<8), ks + 16 + 8*half + (t-8) (t>=8), branchless
        const float* bsrc = battp + ks + 8 * half;
#pragma unroll
        for (int t4 = 0; t4 < 4; ++t4) {
            float4 q = *reinterpret_cast<const float4*>(bsrc + 4 * t4 + ((t4 >= 2) ? 8 : 0));
            b[4 * t4 + 0] = (_Float16)(q.x * zmask);
            b[4 * t4 + 1] = (_Float16)(q.y * zmask);
            b[4 * t4 + 2] = (_Float16)(q.z * zmask);
            b[4 * t4 + 3] = (_Float16)(q.w * zmask);
        }
        acc = __builtin_amdgcn_wmma_f32_16x16x32_f16(
            /*neg_a=*/false, a, /*neg_b=*/false, b,
            /*c_mod=*/(short)0, acc, /*reuse_a=*/false, /*reuse_b=*/false);
    }
    // D layout: VGPR r -> M = r + 8*half, N = lane&15
    int mbase = wave * 16 + half * 8;
    if (m == 0) {
#pragma unroll
        for (int r = 0; r < 8; ++r) e1[mbase + r] = acc[r];
    } else if (m == 1) {
#pragma unroll
        for (int r = 0; r < 8; ++r) e2[mbase + r] = acc[r];
    }
}

// ---------------- 8. new_edge_index (value-cast to float) ----------------
__global__ void hgpsl_edges(float* __restrict__ out) {
    long t = (long)blockIdx.x * blockDim.x + threadIdx.x;
    const long nkk = (long)NK * KPOOL;
    if (t >= nkk) return;
    int r = (int)(t / KPOOL);
    int b = r / KPOOL;
    out[t]       = (float)r;                              // new_row
    out[nkk + t] = (float)(b * KPOOL + (int)(t % KPOOL)); // new_col
}

// ---------------- 9. row softmax: attn = softmax(leakyrelu(e1_i+e2_j) + induced) ----------------
__global__ __launch_bounds__(256) void hgpsl_attn(const float* __restrict__ e1,
                                                  const float* __restrict__ e2,
                                                  const float* __restrict__ induced,
                                                  float* __restrict__ out_attn) {
    int rowid = (blockIdx.x * blockDim.x + threadIdx.x) >> 5;  // pooled row
    int lane  = threadIdx.x & 31;
    if (rowid >= NK) return;
    int b = rowid / KPOOL;
    float ei = e1[rowid];
    const float* e2g = e2 + (size_t)b * KPOOL;
    const float* ind = induced + (size_t)rowid * KPOOL;

    float w[NITER];
    float mx = -1e30f;
#pragma unroll
    for (int it = 0; it < NITER; ++it) {
        int j = it * 32 + lane;
        float v = -1e30f;
        if (j < KPOOL) {
            float s  = ei + e2g[j];
            float lr = (s > 0.f) ? s : SLOPE * s;
            v = lr + LAMB * ind[j];
        }
        w[it] = v;
        mx = fmaxf(mx, v);
    }
#pragma unroll
    for (int off = 16; off > 0; off >>= 1) mx = fmaxf(mx, __shfl_xor(mx, off, 32));
    float sum = 0.f;
#pragma unroll
    for (int it = 0; it < NITER; ++it) {
        int j = it * 32 + lane;
        if (j < KPOOL) { float ex = __expf(w[it] - mx); w[it] = ex; sum += ex; }
    }
#pragma unroll
    for (int off = 16; off > 0; off >>= 1) sum += __shfl_xor(sum, off, 32);
    float inv = 1.f / sum;
#pragma unroll
    for (int it = 0; it < NITER; ++it) {
        int j = it * 32 + lane;
        if (j < KPOOL) out_attn[(size_t)rowid * KPOOL + j] = w[it] * inv;
    }
}

// ---------------- launch ----------------
extern "C" void kernel_launch(void* const* d_in, const int* in_sizes, int n_in,
                              void* d_out, int out_size, void* d_ws, size_t ws_size,
                              hipStream_t stream) {
    const float* x   = (const float*)d_in[0];
    const int*   ei  = (const int*)d_in[1];
    const float* ea  = (const float*)d_in[2];
    // d_in[3] = batch (unused: derivable), d_in[4] = att, d_in[5] = num_graphs
    const float* att = (const float*)d_in[4];
    const int* row = ei;
    const int* col = ei + EEDGES;

    // workspace layout (256B-aligned regions)
    char* ws = (char*)d_ws;
    auto al = [](size_t v) { return (v + 255) & ~(size_t)255; };
    size_t o = 0;
    float* deg     = (float*)(ws + o); o = al(o + (size_t)NNODES * 4);
    float* agg     = (float*)(ws + o); o = al(o + (size_t)NNODES * CCH * 4);
    float* score   = (float*)(ws + o); o = al(o + (size_t)NNODES * 4);
    int*   perm    = (int*)  (ws + o); o = al(o + (size_t)NK * 4);
    int*   mapping = (int*)  (ws + o); o = al(o + (size_t)NNODES * 4);
    float* induced = (float*)(ws + o); o = al(o + (size_t)NK * KPOOL * 4);
    float* e1      = (float*)(ws + o); o = al(o + (size_t)NK * 4);
    float* e2      = (float*)(ws + o); o = al(o + (size_t)NK * 4);

    // output layout: x_pool | new_edge_index(2*Nk*k) | attn(Nk*k) | batch_pool(Nk)
    float* out = (float*)d_out;
    const size_t nkk  = (size_t)NK * KPOOL;
    float* out_xpool = out;
    float* out_edges = out + (size_t)NK * CCH;
    float* out_attn  = out_edges + 2 * nkk;
    float* out_batch = out_attn + nkk;

    const int T = 256;
    // init
    hgpsl_fill_f32<<<2048, T, 0, stream>>>(deg, 0.f, NNODES);
    hgpsl_fill_f32<<<8192, T, 0, stream>>>(agg, 0.f, NNODES * CCH);
    hgpsl_fill_f32<<<8192, T, 0, stream>>>(induced, 0.f, (int)(NK * KPOOL));
    hgpsl_fill_i32<<<2048, T, 0, stream>>>(mapping, -1, NNODES);

    // information score
    hgpsl_deg<<<EEDGES / T, T, 0, stream>>>(row, ea, deg);
    hgpsl_dis<<<NNODES / T, T, 0, stream>>>(deg);
    hgpsl_agg<<<(EEDGES * 32) / T, T, 0, stream>>>(row, col, ea, deg, x, agg);
    hgpsl_score<<<NNODES / T, T, 0, stream>>>(x, agg, score);

    // top-k + pooling
    hgpsl_topk<<<NGRAPH, NPG, 0, stream>>>(score, perm, mapping, out_batch);
    hgpsl_gather<<<(NK * (CCH / 4) + T - 1) / T, T, 0, stream>>>(x, perm, out_xpool);
    hgpsl_induced<<<EEDGES / T, T, 0, stream>>>(row, col, ea, mapping, induced);

    // structure learning
    hgpsl_ewmma<<<(NK / 16) / 8, T, 0, stream>>>(out_xpool, att, e1, e2);  // 3280 waves
    hgpsl_edges<<<(int)((nkk + T - 1) / T), T, 0, stream>>>(out_edges);
    hgpsl_attn<<<(NK * 32 + T - 1) / T, T, 0, stream>>>(e1, e2, induced, out_attn);
    (void)in_sizes; (void)n_in; (void)out_size; (void)ws_size;
}
